// ConditionalFeedForward_89790586290426
// MI455X (gfx1250) — matrix operand order repaired and checked by reading, not compiled, not run on hardware
//
#include <hip/hip_runtime.h>
#include <hip/hip_bf16.h>

// ---------------- problem constants ----------------
#define T_TOK 8192
#define DIM   2560
#define INTER 1664
#define NEXP  16
#define TOPK  6
#define CAP   (T_TOK * TOPK)          // worst-case slots per expert (dupes allowed)
#define MT    2                        // M fragments per wave (32 rows)
#define BM    128                      // block M tile (4 waves * 32 rows)

// gfx1250 async direct-to-LDS path (ASYNCcnt) if the toolchain declares it
#if defined(__AMDGCN__) && \
    __has_builtin(__builtin_amdgcn_global_load_async_to_lds_b128) && \
    __has_builtin(__builtin_amdgcn_s_wait_asynccnt)
#define USE_ASYNC_LDS 1
#else
#define USE_ASYNC_LDS 0
#endif

// ---------------- workspace layout (bytes) ----------------
// counts[16] | offsets[17] | tok[E*CAP] | wgt[E*CAP] | xb bf16 | w1b | w3b | w2b | hbuf
#define WS_COUNTS   ((size_t)0)
#define WS_OFFSETS  ((size_t)256)
#define WS_TOK      ((size_t)4096)
#define SZ_LIST     ((size_t)NEXP * CAP * 4)                 // 3 MB
#define WS_WGT      (WS_TOK + SZ_LIST)
#define WS_XB       (WS_WGT + SZ_LIST)
#define SZ_XB       ((size_t)T_TOK * DIM * 2)                // 42 MB
#define SZ_W        ((size_t)NEXP * INTER * DIM * 2)         // 136 MB each
#define WS_W1B      (WS_XB + SZ_XB)
#define WS_W3B      (WS_W1B + SZ_W)
#define WS_W2B      (WS_W3B + SZ_W)
#define WS_HBUF     (WS_W2B + SZ_W)                          // <=51200 rows * INTER bf16

typedef __attribute__((ext_vector_type(16))) __bf16 v16bf;
typedef __attribute__((ext_vector_type(8)))  __bf16 v8bf;
typedef __attribute__((ext_vector_type(8)))  float  v8f;
typedef __attribute__((ext_vector_type(4)))  float  v4f;
typedef int v4i_gcc __attribute__((vector_size(16)));   // matches builtin prototype

// A operand fragment from a bf16 row: two 16-byte chunks per lane
// (lanes 0-15: K k..k+7 and k+8..k+15 ; lanes 16-31: K k+16..k+23 and k+24..k+31)
__device__ __forceinline__ v16bf load_frag_a(const __bf16* __restrict__ p0,
                                             const __bf16* __restrict__ p1) {
    union { v16bf v; v8bf h[2]; } u;
    u.h[0] = *(const v8bf*)p0;
    u.h[1] = *(const v8bf*)p1;
    return u.v;
}

#define WMMA_BF16(A, B, C) \
    __builtin_amdgcn_wmma_f32_16x16x32_bf16(false, (A), false, (B), (short)0, (C), false, false)

// ---- cooperative 64(N) x 32(K) bf16 tile staging: 256 chunks of 16B, 2/thread ----
#if USE_ASYNC_LDS
__device__ __forceinline__ void async_b128(const void* g, void* l) {
    __builtin_amdgcn_global_load_async_to_lds_b128(
        (__attribute__((address_space(1))) v4i_gcc*)g,
        (__attribute__((address_space(3))) v4i_gcc*)l,
        0, 0);
}
__device__ __forceinline__ void tile_g2lds_async(const __bf16* __restrict__ src,
                                                 size_t stride, int nbase, int k,
                                                 int tid, __bf16* dst) {
#pragma unroll
    for (int i = 0; i < 2; ++i) {
        int c = tid + i * 128;                 // chunk 0..255
        int n = c >> 2;
        int ko = (c & 3) * 8;
        async_b128(src + (size_t)(nbase + n) * stride + k + ko, dst + n * 32 + ko);
    }
}
#define TILE_WAIT() __builtin_amdgcn_s_wait_asynccnt(0)
#else
__device__ __forceinline__ void tile_g2r(const __bf16* __restrict__ src, size_t stride,
                                         int nbase, int k, int tid, v8bf r[2]) {
#pragma unroll
    for (int i = 0; i < 2; ++i) {
        int c = tid + i * 128;
        int n = c >> 2;
        int ko = (c & 3) * 8;
        r[i] = *(const v8bf*)(src + (size_t)(nbase + n) * stride + k + ko);
    }
}
__device__ __forceinline__ void tile_r2lds(__bf16* dst, int tid, const v8bf r[2]) {
#pragma unroll
    for (int i = 0; i < 2; ++i) {
        int c = tid + i * 128;
        int n = c >> 2;
        int ko = (c & 3) * 8;
        *(v8bf*)(dst + n * 32 + ko) = r[i];
    }
}
#endif

// ---------------- setup kernels ----------------
__global__ void k_zero(float* __restrict__ out, int* __restrict__ counts) {
    size_t i = (size_t)blockIdx.x * blockDim.x + threadIdx.x;
    if (i < (size_t)T_TOK * DIM) out[i] = 0.0f;
    if (i < NEXP) counts[i] = 0;
}

__global__ void k_route(const int* __restrict__ idx, const float* __restrict__ ew,
                        int* __restrict__ counts, int* __restrict__ tok,
                        float* __restrict__ wgt) {
    int i = blockIdx.x * blockDim.x + threadIdx.x;
    if (i >= T_TOK * TOPK) return;
    int e = idx[i] & (NEXP - 1);
    int slot = atomicAdd(&counts[e], 1);
    tok[(size_t)e * CAP + slot] = i / TOPK;
    wgt[(size_t)e * CAP + slot] = ew[i];
}

__global__ void k_scan(const int* __restrict__ counts, int* __restrict__ offsets) {
    if (threadIdx.x == 0 && blockIdx.x == 0) {
        int acc = 0;
        for (int e = 0; e < NEXP; ++e) {
            offsets[e] = acc;
            acc += (counts[e] + (BM - 1)) & ~(BM - 1);  // pad expert to 128-row block tiles
        }
        offsets[NEXP] = acc;
    }
}

// generic fp32 -> bf16 repack, 8 elements/thread (all sizes are multiples of 8)
__global__ void k_f32_to_bf16(const float* __restrict__ src, __bf16* __restrict__ dst,
                              size_t n) {
    size_t i = ((size_t)blockIdx.x * blockDim.x + threadIdx.x) * 8;
    if (i >= n) return;
    v4f a = *(const v4f*)(src + i);
    v4f b = *(const v4f*)(src + i + 4);
    v8bf r;
#pragma unroll
    for (int j = 0; j < 4; ++j) {
        r[j]     = (__bf16)a[j];
        r[4 + j] = (__bf16)b[j];
    }
    *(v8bf*)(dst + i) = r;
}

// ---------------- stage 1: h = silu(x W1^T) * (x W3^T), permuted rows ----------------
// grid: (CAP/128, INTER/64, NEXP), block: 128 (4 waves); block tile 128x64 x {w1,w3};
// weight tiles double-buffered in LDS (async DMA), shared by all 4 waves.
__global__ __launch_bounds__(128)
void k_stage1(const __bf16* __restrict__ w1b, const __bf16* __restrict__ w3b,
              const __bf16* __restrict__ xb, const int* __restrict__ counts,
              const int* __restrict__ offsets, const int* __restrict__ tok,
              __bf16* __restrict__ hbuf) {
    const int e = blockIdx.z;
    const int cnt = counts[e];
    const int mtile = blockIdx.x;
    if (mtile * BM >= cnt) return;          // block-uniform exit (same cnt everywhere)
    const int nbase = blockIdx.y * 64;
    const int tid  = threadIdx.x;
    const int wave = tid >> 5;
    const int lane = tid & 31;
    const int half = lane >> 4;
    const int lm   = lane & 15;
    const int wbase = mtile * BM + wave * (MT * 16);

    __shared__ __align__(64) __bf16 sb1[2][64 * 32];   // [buf][n][k] 4KB each
    __shared__ __align__(64) __bf16 sb3[2][64 * 32];

    // gather token rows for the two A fragments (clamp tail; masked in stage 2)
    const __bf16* xrow[MT];
#pragma unroll
    for (int m = 0; m < MT; ++m) {
        int r = wbase + m * 16 + lm;
        if (r >= cnt) r = cnt - 1;
        xrow[m] = xb + (size_t)tok[(size_t)e * CAP + r] * DIM;
    }

    const __bf16* __restrict__ w1e = w1b + (size_t)e * INTER * DIM;
    const __bf16* __restrict__ w3e = w3b + (size_t)e * INTER * DIM;

    v8f acc1[MT][4];
    v8f acc3[MT][4];
#pragma unroll
    for (int m = 0; m < MT; ++m)
#pragma unroll
        for (int s = 0; s < 4; ++s) { acc1[m][s] = v8f{}; acc3[m][s] = v8f{}; }

    // prologue: stage first K tile
#if USE_ASYNC_LDS
    tile_g2lds_async(w1e, DIM, nbase, 0, tid, sb1[0]);
    tile_g2lds_async(w3e, DIM, nbase, 0, tid, sb3[0]);
    TILE_WAIT();
#else
    {
        v8bf r1[2], r3[2];
        tile_g2r(w1e, DIM, nbase, 0, tid, r1);
        tile_g2r(w3e, DIM, nbase, 0, tid, r3);
        tile_r2lds(sb1[0], tid, r1);
        tile_r2lds(sb3[0], tid, r3);
    }
#endif
    __syncthreads();

    int cur = 0;
    for (int k = 0; k < DIM; k += 32) {
        const bool has_next = (k + 32) < DIM;
#if USE_ASYNC_LDS
        if (has_next) {                        // DMA next tile while computing
            tile_g2lds_async(w1e, DIM, nbase, k + 32, tid, sb1[cur ^ 1]);
            tile_g2lds_async(w3e, DIM, nbase, k + 32, tid, sb3[cur ^ 1]);
        }
#else
        v8bf r1[2], r3[2];
        if (has_next) {
            tile_g2r(w1e, DIM, nbase, k + 32, tid, r1);
            tile_g2r(w3e, DIM, nbase, k + 32, tid, r3);
        }
#endif

        v16bf a[MT];
#pragma unroll
        for (int m = 0; m < MT; ++m)
            a[m] = load_frag_a(xrow[m] + k + half * 16, xrow[m] + k + half * 16 + 8);

#pragma unroll
        for (int s = 0; s < 4; ++s) {
            const int bo = (s * 16 + lm) * 32 + half * 16;
            v16bf b1 = *(const v16bf*)(&sb1[cur][bo]);   // 2x ds_load_b128
            v16bf b3 = *(const v16bf*)(&sb3[cur][bo]);
#pragma unroll
            for (int m = 0; m < MT; ++m) {
                acc1[m][s] = WMMA_BF16(a[m], b1, acc1[m][s]);
                acc3[m][s] = WMMA_BF16(a[m], b3, acc3[m][s]);
            }
        }

#if USE_ASYNC_LDS
        if (has_next) TILE_WAIT();
#else
        if (has_next) {
            tile_r2lds(sb1[cur ^ 1], tid, r1);
            tile_r2lds(sb3[cur ^ 1], tid, r3);
        }
#endif
        __syncthreads();
        cur ^= 1;
    }

    // epilogue: SwiGLU, write bf16 into permuted h buffer
    const int rowbase = offsets[e] + wbase;
#pragma unroll
    for (int m = 0; m < MT; ++m) {
#pragma unroll
        for (int s = 0; s < 4; ++s) {
#pragma unroll
            for (int v = 0; v < 8; ++v) {
                float g = acc1[m][s][v];
                float h = (g / (1.0f + __expf(-g))) * acc3[m][s][v];
                int M = m * 16 + v + half * 8;   // C/D layout: VGPR v, half selects M 0-7/8-15
                hbuf[(size_t)(rowbase + M) * INTER + nbase + s * 16 + lm] = (__bf16)h;
            }
        }
    }
}

// ---------------- stage 2: out[t] += score * (h W2^T) ----------------
// grid: (CAP/128, DIM/64, NEXP), block: 128 (4 waves); w2 tile double-buffered in LDS
__global__ __launch_bounds__(128)
void k_stage2(const __bf16* __restrict__ w2b, const __bf16* __restrict__ hbuf,
              const int* __restrict__ counts, const int* __restrict__ offsets,
              const int* __restrict__ tok, const float* __restrict__ wgt,
              float* __restrict__ out) {
    const int e = blockIdx.z;
    const int cnt = counts[e];
    const int mtile = blockIdx.x;
    if (mtile * BM >= cnt) return;
    const int nbase = blockIdx.y * 64;
    const int tid  = threadIdx.x;
    const int wave = tid >> 5;
    const int lane = tid & 31;
    const int half = lane >> 4;
    const int lm   = lane & 15;
    const int wbase = mtile * BM + wave * (MT * 16);

    __shared__ __align__(64) __bf16 sb[2][64 * 32];

    const int rowbase = offsets[e] + wbase;
    const __bf16* arow[MT];
#pragma unroll
    for (int m = 0; m < MT; ++m)
        arow[m] = hbuf + (size_t)(rowbase + m * 16 + lm) * INTER;   // rows pre-permuted

    const __bf16* __restrict__ w2e = w2b + (size_t)e * DIM * INTER;

    v8f acc[MT][4];
#pragma unroll
    for (int m = 0; m < MT; ++m)
#pragma unroll
        for (int s = 0; s < 4; ++s) acc[m][s] = v8f{};

#if USE_ASYNC_LDS
    tile_g2lds_async(w2e, INTER, nbase, 0, tid, sb[0]);
    TILE_WAIT();
#else
    {
        v8bf r[2];
        tile_g2r(w2e, INTER, nbase, 0, tid, r);
        tile_r2lds(sb[0], tid, r);
    }
#endif
    __syncthreads();

    int cur = 0;
    for (int k = 0; k < INTER; k += 32) {
        const bool has_next = (k + 32) < INTER;
#if USE_ASYNC_LDS
        if (has_next) tile_g2lds_async(w2e, INTER, nbase, k + 32, tid, sb[cur ^ 1]);
#else
        v8bf r[2];
        if (has_next) tile_g2r(w2e, INTER, nbase, k + 32, tid, r);
#endif

        v16bf a[MT];
#pragma unroll
        for (int m = 0; m < MT; ++m)
            a[m] = load_frag_a(arow[m] + k + half * 16, arow[m] + k + half * 16 + 8);

#pragma unroll
        for (int s = 0; s < 4; ++s) {
            v16bf b = *(const v16bf*)(&sb[cur][(s * 16 + lm) * 32 + half * 16]);
#pragma unroll
            for (int m = 0; m < MT; ++m)
                acc[m][s] = WMMA_BF16(a[m], b, acc[m][s]);
        }

#if USE_ASYNC_LDS
        if (has_next) TILE_WAIT();
#else
        if (has_next) tile_r2lds(sb[cur ^ 1], tid, r);
#endif
        __syncthreads();
        cur ^= 1;
    }

    // epilogue: scale by router weight, scatter-add into out (atomics combine top-k)
#pragma unroll
    for (int m = 0; m < MT; ++m) {
#pragma unroll
        for (int v = 0; v < 8; ++v) {
            int row = wbase + m * 16 + v + half * 8;
            if (row < cnt) {
                int tt = tok[(size_t)e * CAP + row];
                float sc = wgt[(size_t)e * CAP + row];
#pragma unroll
                for (int s = 0; s < 4; ++s)
                    atomicAdd(&out[(size_t)tt * DIM + nbase + s * 16 + lm],
                              sc * acc[m][s][v]);
            }
        }
    }
}

// ---------------- launch ----------------
extern "C" void kernel_launch(void* const* d_in, const int* in_sizes, int n_in,
                              void* d_out, int out_size, void* d_ws, size_t ws_size,
                              hipStream_t stream) {
    const float* x   = (const float*)d_in[0];
    const int*   idx = (const int*)d_in[1];
    const float* ew  = (const float*)d_in[2];
    const float* w1  = (const float*)d_in[3];
    const float* w2  = (const float*)d_in[4];
    const float* w3  = (const float*)d_in[5];
    float* out = (float*)d_out;

    char* ws = (char*)d_ws;
    int*     counts  = (int*)(ws + WS_COUNTS);
    int*     offsets = (int*)(ws + WS_OFFSETS);
    int*     tok     = (int*)(ws + WS_TOK);
    float*   wgt     = (float*)(ws + WS_WGT);
    __bf16*  xb      = (__bf16*)(ws + WS_XB);
    __bf16*  w1b     = (__bf16*)(ws + WS_W1B);
    __bf16*  w3b     = (__bf16*)(ws + WS_W3B);
    __bf16*  w2b     = (__bf16*)(ws + WS_W2B);
    __bf16*  hbuf    = (__bf16*)(ws + WS_HBUF);

    {   // zero out + counts
        size_t n = (size_t)T_TOK * DIM;
        k_zero<<<dim3((unsigned)((n + 255) / 256)), dim3(256), 0, stream>>>(out, counts);
    }
    k_route<<<dim3((T_TOK * TOPK + 255) / 256), dim3(256), 0, stream>>>(idx, ew, counts, tok, wgt);
    k_scan<<<dim3(1), dim3(32), 0, stream>>>(counts, offsets);

    // one-time bf16 repacks (bandwidth-bound, ~55us total at HBM speed)
    {
        size_t nx = (size_t)T_TOK * DIM;
        size_t nw = (size_t)NEXP * INTER * DIM;
        k_f32_to_bf16<<<dim3((unsigned)((nx / 8 + 255) / 256)), dim3(256), 0, stream>>>(x, xb, nx);
        k_f32_to_bf16<<<dim3((unsigned)((nw / 8 + 255) / 256)), dim3(256), 0, stream>>>(w1, w1b, nw);
        k_f32_to_bf16<<<dim3((unsigned)((nw / 8 + 255) / 256)), dim3(256), 0, stream>>>(w3, w3b, nw);
        k_f32_to_bf16<<<dim3((unsigned)((nw / 8 + 255) / 256)), dim3(256), 0, stream>>>(w2, w2b, nw);
    }
    {
        dim3 g1(CAP / BM, INTER / 64, NEXP);
        k_stage1<<<g1, dim3(128), 0, stream>>>(w1b, w3b, xb, counts, offsets, tok, hbuf);
    }
    {
        dim3 g2(CAP / BM, DIM / 64, NEXP);
        k_stage2<<<g2, dim3(128), 0, stream>>>(w2b, hbuf, counts, offsets, tok, wgt, out);
    }
}
